// Decoder_850403525057
// MI455X (gfx1250) — matrix operand order repaired and checked by reading, not compiled, not run on hardware
//
#include <hip/hip_runtime.h>
#include <hip/hip_bf16.h>
#include <math.h>
#include <stdint.h>

// MI455X / gfx1250, wave32. bf16 WMMA path: v_wmma_f32_16x16x32_bf16,
// TDM async xp staging: tensor_load_to_lds + s_wait_tensorcnt.

typedef __attribute__((ext_vector_type(16))) __bf16 v16bf;
typedef __attribute__((ext_vector_type(8)))  __bf16 v8bf;
typedef __attribute__((ext_vector_type(8)))  float  v8f;
typedef __attribute__((ext_vector_type(4)))  float  v4f;
typedef __attribute__((ext_vector_type(4)))  unsigned int u32x4;
typedef __attribute__((ext_vector_type(4)))  int    i32x4;
typedef __attribute__((ext_vector_type(8)))  int    i32x8;

#if defined(__gfx1250__) && __has_builtin(__builtin_amdgcn_tensor_load_to_lds)
#define HAVE_TDM 1
#else
#define HAVE_TDM 0
#endif

// ---------------------------------------------------------------------------
// Fragment helpers (wave32). l = lane&15, hi = lane>>4.
//   A (16x32 bf16 from row-major MxK): halfs 0..7  = A[m0+l][k0+hi*8 + j]
//                                      halfs 8..15 = A[m0+l][k0+16+hi*8 + j]
//   B (32x16 bf16 from row-major NxK W): halfs 0..7  = W[n0+l][k0+hi*8 + j]
//                                        halfs 8..15 = W[n0+l][k0+16+hi*8 + j]
//   C/D f32: element r -> (m0 + r + 8*hi, n0 + l)
// ---------------------------------------------------------------------------

__device__ __forceinline__ v8f wmma_bf16(v16bf a, v16bf b, v8f c) {
    return __builtin_amdgcn_wmma_f32_16x16x32_bf16(
        false, a, false, b, (short)0, c, false, false);
}

__device__ __forceinline__ v16bf load_a_f32(const float* __restrict__ Arow,
                                            int k0, int hi) {
    const v4f* p0 = (const v4f*)(Arow + k0 + hi * 8);
    const v4f* p1 = (const v4f*)(Arow + k0 + 16 + hi * 8);
    v4f a0 = p0[0], a1 = p0[1];
    v4f b0 = p1[0], b1 = p1[1];
    v16bf a;
#pragma unroll
    for (int j = 0; j < 4; ++j) {
        a[j]      = (__bf16)a0[j];
        a[j + 4]  = (__bf16)a1[j];
        a[j + 8]  = (__bf16)b0[j];
        a[j + 12] = (__bf16)b1[j];
    }
    return a;
}

__device__ __forceinline__ v16bf load_a_bf16(const __bf16* Arow, int k0, int hi) {
    const v8bf* p0 = (const v8bf*)(Arow + k0 + hi * 8);
    const v8bf* p1 = (const v8bf*)(Arow + k0 + 16 + hi * 8);
    v8bf lo = p0[0], hh = p1[0];
    v16bf a;
#pragma unroll
    for (int j = 0; j < 8; ++j) { a[j] = lo[j]; a[j + 8] = hh[j]; }
    return a;
}

__device__ __forceinline__ v16bf load_b_bf16(const __bf16* __restrict__ Wrow,
                                             int k0, int hi) {
    const v8bf* p0 = (const v8bf*)(Wrow + k0 + hi * 8);
    const v8bf* p1 = (const v8bf*)(Wrow + k0 + 16 + hi * 8);
    v8bf lo = p0[0], hh = p1[0];
    v16bf b;
#pragma unroll
    for (int j = 0; j < 8; ++j) { b[j] = lo[j]; b[j + 8] = hh[j]; }
    return b;
}

// ---------------------------------------------------------------------------
// Kernel 0: f32 -> bf16 weight repack (row-major preserved)
// ---------------------------------------------------------------------------
__global__ void pack_bf16_kernel(const float* __restrict__ src,
                                 __bf16* __restrict__ dst, int n) {
    int i = blockIdx.x * blockDim.x + threadIdx.x;
    if (i < n) dst[i] = (__bf16)src[i];
}

// ---------------------------------------------------------------------------
// Kernel 1: xp = x @ W_ih^T + b_ih + b_hh     (32768 x 256) x (256 -> 512)
// Two n-tiles per wave sharing one A fragment (dual WMMA chains hide the
// bf16 WMMA hazard slots and halve conversion work).
// ---------------------------------------------------------------------------
__global__ __launch_bounds__(256) void xp_kernel(
    const float* __restrict__ x,      // (B*S, 256) f32
    const __bf16* __restrict__ Wihb,  // (512, 256) bf16 row-major
    const float* __restrict__ b_ih,
    const float* __restrict__ b_hh,
    float* __restrict__ XP)           // (B*S, 512) f32
{
    const int wave = threadIdx.x >> 5;
    const int lane = threadIdx.x & 31;
    const int l = lane & 15, hi = lane >> 4;
    const int unit = blockIdx.x * 8 + wave;   // 32768 units: mt(2048) x p(16)
    const int mt = unit >> 4, p = unit & 15;
    const int m0 = mt * 16;
    const int n_a = p * 32 + l, n_b = n_a + 16;

    v8f c0, c1;
    const float bias0 = b_ih[n_a] + b_hh[n_a];
    const float bias1 = b_ih[n_b] + b_hh[n_b];
#pragma unroll
    for (int r = 0; r < 8; ++r) { c0[r] = bias0; c1[r] = bias1; }

    const float* Arow  = x + (size_t)(m0 + l) * 256;
    const __bf16* W0   = Wihb + (size_t)n_a * 256;
    const __bf16* W1   = Wihb + (size_t)n_b * 256;

#pragma unroll
    for (int kc = 0; kc < 8; ++kc) {
        v16bf a  = load_a_f32(Arow, kc * 32, hi);
        v16bf w0 = load_b_bf16(W0, kc * 32, hi);
        v16bf w1 = load_b_bf16(W1, kc * 32, hi);
        c0 = wmma_bf16(a, w0, c0);
        c1 = wmma_bf16(a, w1, c1);
    }

    float* o = XP + (size_t)(m0 + 8 * hi) * 512;
#pragma unroll
    for (int r = 0; r < 8; ++r) {
        o[(size_t)r * 512 + n_a] = c0[r];
        o[(size_t)r * 512 + n_b] = c1[r];
    }
}

// ---------------------------------------------------------------------------
// TDM descriptor: load a 16-row x 512-col f32 tile (rows strided by
// 512*512 elements in XP) into LDS at lds_off. Packed per ISA 08 §8.3/8.4.
// ---------------------------------------------------------------------------
#if HAVE_TDM
__device__ __forceinline__ void tdm_load_xp_tile(const float* gptr,
                                                 unsigned int lds_off) {
    uint64_t ga = (uint64_t)(uintptr_t)gptr;
    u32x4 g0;
    g0[0] = 1u;                                   // count=1, user descriptor
    g0[1] = lds_off;                              // lds_addr (bytes)
    g0[2] = (unsigned int)ga;                     // global_addr[31:0]
    g0[3] = (unsigned int)((ga >> 32) & 0x1FFFFFFu) | (2u << 30); // addr hi | type=2
    i32x8 g1;
    g1[0] = (int)(2u << 16);          // wg_mask=0, data_size=2 (4 bytes)
    g1[1] = (int)(512u << 16);        // tensor_dim0[15:0]=512 in bits[63:48]
    g1[2] = (int)(16u << 16);         // tensor_dim1[15:0]=16 in bits[95:80]
    g1[3] = (int)(512u << 16);        // tile_dim0=512 in bits[127:112]
    g1[4] = 16;                       // tile_dim1=16 in bits[143:128]
    g1[5] = 512 * 512;                // tensor_dim0_stride[31:0]
    g1[6] = 0;
    g1[7] = 0;
    i32x4 z4 = {0, 0, 0, 0};
#if __clang_major__ >= 23
    i32x8 z8 = {0, 0, 0, 0, 0, 0, 0, 0};
    __builtin_amdgcn_tensor_load_to_lds(g0, g1, z4, z4, z8, 0);
#else
    __builtin_amdgcn_tensor_load_to_lds(g0, g1, z4, z4, 0);
#endif
}
#endif

// ---------------------------------------------------------------------------
// Kernel 2: recurrent scan. 4 blocks x 16 waves; block owns 16 batch rows.
// h double-buffered in LDS (bf16). xp tiles double-buffered in LDS via the
// Tensor Data Mover (issued by wave 0, overlapped with the WMMA step).
// ---------------------------------------------------------------------------
__global__ __launch_bounds__(512) void rnn_scan_kernel(
    const float* __restrict__ h0,     // (64, 512) f32
    const __bf16* __restrict__ Whhb,  // (512, 512) bf16 row-major
    float* __restrict__ XP)           // (B*S, 512): xp in, hs out (in place)
{
    __shared__ __bf16 hbuf[2][16 * 512];   // 32 KB
#if HAVE_TDM
    __shared__ float xpbuf[2][16 * 512];   // 64 KB (TDM double buffer)
#endif

    const int wave = threadIdx.x >> 5;     // 0..15
    const int lane = threadIdx.x & 31;
    const int l = lane & 15, hi = lane >> 4;
    const int b0 = blockIdx.x * 16;

    const size_t rstep = (size_t)512 * 512;            // +1 batch row in XP

#if HAVE_TDM
    if (wave == 0) {
        tdm_load_xp_tile(XP + (size_t)b0 * rstep,
                         (unsigned int)(uintptr_t)&xpbuf[0][0]);
    }
#endif

    // Load h0 into buffer 0 (bf16, row-major 16x512).
    for (int i = threadIdx.x; i < 16 * 512; i += 512)
        hbuf[0][i] = (__bf16)h0[(size_t)(b0 + (i >> 9)) * 512 + (i & 511)];
#if HAVE_TDM
    if (wave == 0) __builtin_amdgcn_s_wait_tensorcnt(0);
#endif
    __syncthreads();

    // This wave owns two adjacent n-tiles: columns n_a and n_b.
    const int n_a = wave * 32 + l;
    const int n_b = n_a + 16;
    const __bf16* Wrow0 = Whhb + (size_t)n_a * 512;
    const __bf16* Wrow1 = Whhb + (size_t)n_b * 512;

    int cur = 0;
    for (int t = 0; t < 512; ++t) {
        const int nxt = cur ^ 1;
        // XP element for C-frag slot r: row (b0 + r + 8*hi), time t.
        const size_t base = ((size_t)(b0 + 8 * hi) * 512 + (size_t)t) * 512;

#if HAVE_TDM
        // Kick off the DMA of next step's xp tile while we compute.
        if (wave == 0 && t + 1 < 512) {
            tdm_load_xp_tile(XP + (size_t)b0 * rstep + (size_t)(t + 1) * 512,
                             (unsigned int)(uintptr_t)&xpbuf[nxt][0]);
        }
#endif

        v8f c0, c1;
#pragma unroll
        for (int r = 0; r < 8; ++r) {
#if HAVE_TDM
            c0[r] = xpbuf[cur][(r + 8 * hi) * 512 + n_a];
            c1[r] = xpbuf[cur][(r + 8 * hi) * 512 + n_b];
#else
            c0[r] = XP[base + (size_t)r * rstep + n_a];
            c1[r] = XP[base + (size_t)r * rstep + n_b];
#endif
        }
#if !HAVE_TDM
        if (t + 1 < 512) __builtin_prefetch(&XP[base + 512 + n_a], 0, 3);
#endif

        const __bf16* hrow = &hbuf[cur][(size_t)l * 512];
#pragma unroll
        for (int kc = 0; kc < 16; ++kc) {
            v16bf a  = load_a_bf16(hrow, kc * 32, hi);
            v16bf w0 = load_b_bf16(Wrow0, kc * 32, hi);
            v16bf w1 = load_b_bf16(Wrow1, kc * 32, hi);
            c0 = wmma_bf16(a, w0, c0);
            c1 = wmma_bf16(a, w1, c1);
        }

#pragma unroll
        for (int r = 0; r < 8; ++r) {
            float v0 = tanhf(c0[r]);
            float v1 = tanhf(c1[r]);
            XP[base + (size_t)r * rstep + n_a] = v0;   // hs overwrites xp
            XP[base + (size_t)r * rstep + n_b] = v1;
            hbuf[nxt][(r + 8 * hi) * 512 + n_a] = (__bf16)v0;
            hbuf[nxt][(r + 8 * hi) * 512 + n_b] = (__bf16)v1;
        }

#if HAVE_TDM
        if (wave == 0 && t + 1 < 512) __builtin_amdgcn_s_wait_tensorcnt(0);
#endif
        __syncthreads();
        cur = nxt;
    }
}

// ---------------------------------------------------------------------------
// Kernel 3: out = hs @ W_ff^T + b_ff          (32768 x 512) x (512 -> 256)
// Two n-tiles per wave sharing one A fragment.
// ---------------------------------------------------------------------------
__global__ __launch_bounds__(256) void out_kernel(
    const float* __restrict__ HS,     // (B*S, 512) f32 (the in-place XP buffer)
    const __bf16* __restrict__ Wffb,  // (256, 512) bf16 row-major
    const float* __restrict__ b_ff,
    float* __restrict__ out)          // (B*S, 256) f32
{
    const int wave = threadIdx.x >> 5;
    const int lane = threadIdx.x & 31;
    const int l = lane & 15, hi = lane >> 4;
    const int unit = blockIdx.x * 8 + wave;   // 16384 units: mt(2048) x p(8)
    const int mt = unit >> 3, p = unit & 7;
    const int m0 = mt * 16;
    const int n_a = p * 32 + l, n_b = n_a + 16;

    v8f c0, c1;
    const float bias0 = b_ff[n_a];
    const float bias1 = b_ff[n_b];
#pragma unroll
    for (int r = 0; r < 8; ++r) { c0[r] = bias0; c1[r] = bias1; }

    const float* Arow = HS + (size_t)(m0 + l) * 512;
    const __bf16* W0  = Wffb + (size_t)n_a * 512;
    const __bf16* W1  = Wffb + (size_t)n_b * 512;

#pragma unroll
    for (int kc = 0; kc < 16; ++kc) {
        v16bf a  = load_a_f32(Arow, kc * 32, hi);
        v16bf w0 = load_b_bf16(W0, kc * 32, hi);
        v16bf w1 = load_b_bf16(W1, kc * 32, hi);
        c0 = wmma_bf16(a, w0, c0);
        c1 = wmma_bf16(a, w1, c1);
    }

    float* o = out + (size_t)(m0 + 8 * hi) * 256;
#pragma unroll
    for (int r = 0; r < 8; ++r) {
        o[(size_t)r * 256 + n_a] = c0[r];
        o[(size_t)r * 256 + n_b] = c1[r];
    }
}

// ---------------------------------------------------------------------------
// Launch. Workspace layout (bytes):
//   [0, 64MB)      : XP f32 (B*S*512) — xp, then hs in place
//   [64MB, +256KB) : W_ih bf16
//   [.., +512KB)   : W_hh bf16
//   [.., +256KB)   : W_ff bf16     (total ~65 MB)
// ---------------------------------------------------------------------------
extern "C" void kernel_launch(void* const* d_in, const int* in_sizes, int n_in,
                              void* d_out, int out_size, void* d_ws, size_t ws_size,
                              hipStream_t stream) {
    const float* x   = (const float*)d_in[0];   // (64,512,256)
    const float* h0  = (const float*)d_in[1];   // (64,512)
    const float* Wih = (const float*)d_in[2];   // (512,256)
    const float* Whh = (const float*)d_in[3];   // (512,512)
    const float* bih = (const float*)d_in[4];   // (512,)
    const float* bhh = (const float*)d_in[5];   // (512,)
    const float* Wff = (const float*)d_in[6];   // (256,512)
    const float* bff = (const float*)d_in[7];   // (256,)
    float* out = (float*)d_out;

    char* ws = (char*)d_ws;
    float*  XP   = (float*)(ws);
    __bf16* Wihb = (__bf16*)(ws + 67108864);
    __bf16* Whhb = (__bf16*)(ws + 67108864 + 262144);
    __bf16* Wffb = (__bf16*)(ws + 67108864 + 262144 + 524288);

    pack_bf16_kernel<<<512, 256, 0, stream>>>(Wih, Wihb, 512 * 256);
    pack_bf16_kernel<<<1024, 256, 0, stream>>>(Whh, Whhb, 512 * 512);
    pack_bf16_kernel<<<512, 256, 0, stream>>>(Wff, Wffb, 256 * 512);

    xp_kernel<<<4096, 256, 0, stream>>>(x, Wihb, bih, bhh, XP);
    rnn_scan_kernel<<<4, 512, 0, stream>>>(h0, Whhb, XP);
    out_kernel<<<2048, 256, 0, stream>>>(XP, Wffb, bff, out);
}
